// SphereNet_13469017440648
// MI455X (gfx1250) — compile-verified, matrix operand order checked by reading, not run on hardware
//
#include <hip/hip_runtime.h>

// ---------------- problem constants ----------------
#define NEDGE 262144
#define NTRIP 524288

#define MODE_BIAS  1
#define MODE_SWISH 2
#define MODE_MUL   4
#define MODE_ADD   8

typedef __attribute__((ext_vector_type(16))) __bf16 v16bf;
typedef __attribute__((ext_vector_type(8)))  float  v8f;

__device__ __forceinline__ float swishf(float x) {
    return x / (1.0f + __expf(-x));
}

__device__ __forceinline__ void atomAddF(float* p, float v) {
    __hip_atomic_fetch_add(p, v, __ATOMIC_RELAXED, __HIP_MEMORY_SCOPE_AGENT);
}

// =====================================================================
// Weight pre-swizzle: W[K,N] f32 -> fragment-ordered bf16.
// Fragment order (CDNA5 ISA 7.12.2, 16-bit B 32x16 per 32-K chunk):
//   lane L holds col n = ct*16 + (L&15); elems j<8 -> kk = half+j,
//   j>=8 -> kk = 16+half+(j-8), half = (L>>4)*8.
// Output element ((kc*NCT + ct)*32 + lane)*16 + j. K zero-padded to 32.
// =====================================================================
__global__ void swizzle_w(const float* __restrict__ W, __bf16* __restrict__ out,
                          int K, int N, int nkc, int nct)
{
    const int tid = blockIdx.x * 256 + threadIdx.x;     // nkc*nct*32 threads
    if (tid >= nkc * nct * 32) return;
    const int lane = tid & 31;
    const int t2   = tid >> 5;                          // kc*nct + ct
    const int kc   = t2 / nct;
    const int ct   = t2 - kc * nct;
    const int half = (lane >> 4) << 3;
    const int n    = (ct << 4) + (lane & 15);
    v16bf v;
#pragma unroll
    for (int j = 0; j < 8; ++j) {
        int k = (kc << 5) + half + j;
        v[j] = (__bf16)((k < K) ? W[(size_t)k * N + n] : 0.0f);
    }
#pragma unroll
    for (int j = 0; j < 8; ++j) {
        int k = (kc << 5) + 16 + half + j;
        v[8 + j] = (__bf16)((k < K) ? W[(size_t)k * N + n] : 0.0f);
    }
    *(v16bf*)(out + (size_t)tid * 16) = v;
}

// W_bil[i,j,l] (64x64x64) -> fragment-ordered bf16, viewed as a
// [4096 x 64] B matrix with k = j*64+l, chunked by 32 (128 chunks).
__global__ void swizzle_bil(const float* __restrict__ Wb, __bf16* __restrict__ out)
{
    const int tid  = blockIdx.x * 256 + threadIdx.x;    // 128*4*32 = 16384 threads
    const int lane = tid & 31;
    const int t2   = tid >> 5;
    const int kc   = t2 >> 2;
    const int ct   = t2 & 3;
    const int half = (lane >> 4) << 3;
    const int i    = (ct << 4) + (lane & 15);
    const int j    = kc >> 1;
    const int lb   = (kc & 1) << 5;
    v16bf v;
#pragma unroll
    for (int q = 0; q < 8; ++q)
        v[q] = (__bf16)Wb[((size_t)i << 12) + (j << 6) + lb + half + q];
#pragma unroll
    for (int q = 0; q < 8; ++q)
        v[8 + q] = (__bf16)Wb[((size_t)i << 12) + (j << 6) + lb + 16 + half + q];
    *(v16bf*)(out + (size_t)tid * 16) = v;
}

// t [T,294] f32 -> [T,320] bf16 zero-padded (one block per row)
__global__ void cvt_pad_t(const float* __restrict__ t, __bf16* __restrict__ out)
{
    const size_t w = blockIdx.x;
    for (int i = threadIdx.x; i < 320; i += 256)
        out[w * 320 + i] = (__bf16)((i < 294) ? t[w * 294 + i] : 0.0f);
}

// =====================================================================
// Dense GEMM, no LDS: Y[M,N] = epilogue(X[M,K] @ W[K,N]).
// B from fragment-ordered bf16 (cache-resident), A loaded per-lane
// directly from global in fragment order. K must be a multiple of 32.
// Block = 256 threads (8 wave32) covering 128 rows.
// =====================================================================
template <int N, bool ABF16>
__global__ __launch_bounds__(256) void gemm2(
    const void* __restrict__ Xv, const __bf16* __restrict__ Wf,
    const float* __restrict__ bias, const float* __restrict__ auxm,
    const float* __restrict__ auxa, float* __restrict__ Y,
    int K, int mode)
{
    constexpr int NCT = N / 16;
    const int tid  = threadIdx.x;
    const int wave = tid >> 5;
    const int lane = tid & 31;
    const size_t r0   = (size_t)blockIdx.x * 128;
    const size_t rowl = r0 + (wave << 4) + (lane & 15);
    const int half = (lane >> 4) << 3;

    v8f acc[NCT] = {};
    const int nk = K >> 5;

    for (int kc = 0; kc < nk; ++kc) {
        const int kb = kc << 5;
        v16bf a;
        if constexpr (ABF16) {
            const __bf16* xr = (const __bf16*)Xv + rowl * (size_t)K;
            union { uint4 u[2]; v16bf v; } ua;
            ua.u[0] = *(const uint4*)(xr + kb + half);
            ua.u[1] = *(const uint4*)(xr + kb + 16 + half);
            a = ua.v;
            if (kc + 1 < nk) __builtin_prefetch(xr + kb + 32 + half, 0, 0);
        } else {
            const float* xr = (const float*)Xv + rowl * (size_t)K;
            float4 p0 = *(const float4*)(xr + kb + half);
            float4 p1 = *(const float4*)(xr + kb + half + 4);
            float4 p2 = *(const float4*)(xr + kb + 16 + half);
            float4 p3 = *(const float4*)(xr + kb + 16 + half + 4);
            a[0]  = (__bf16)p0.x; a[1]  = (__bf16)p0.y; a[2]  = (__bf16)p0.z; a[3]  = (__bf16)p0.w;
            a[4]  = (__bf16)p1.x; a[5]  = (__bf16)p1.y; a[6]  = (__bf16)p1.z; a[7]  = (__bf16)p1.w;
            a[8]  = (__bf16)p2.x; a[9]  = (__bf16)p2.y; a[10] = (__bf16)p2.z; a[11] = (__bf16)p2.w;
            a[12] = (__bf16)p3.x; a[13] = (__bf16)p3.y; a[14] = (__bf16)p3.z; a[15] = (__bf16)p3.w;
            if (kc + 1 < nk) __builtin_prefetch(xr + kb + 32 + half, 0, 0);
        }
        const __bf16* wf = Wf + ((size_t)(kc * NCT) << 9);   // kc*NCT*512
#pragma unroll
        for (int ct = 0; ct < NCT; ++ct) {
            v16bf b = *(const v16bf*)(wf + (ct << 9) + (lane << 4));
            acc[ct] = __builtin_amdgcn_wmma_f32_16x16x32_bf16(
                false, a, false, b, (short)0, acc[ct], false, false);
        }
    }

    const int hi = (lane >> 4) << 3;
#pragma unroll
    for (int ct = 0; ct < NCT; ++ct) {
        const int col = (ct << 4) + (lane & 15);
#pragma unroll
        for (int r = 0; r < 8; ++r) {
            const size_t row = r0 + (wave << 4) + hi + r;
            float v = acc[ct][r];
            if (mode & MODE_BIAS)  v += bias[col];
            if (mode & MODE_SWISH) v = swishf(v);
            const size_t off = row * (size_t)N + col;
            if (mode & MODE_MUL)   v *= auxm[off];
            if (mode & MODE_ADD)   v += auxa[off];
            Y[off] = v;
        }
    }
}

// =====================================================================
// Fused bilinear einsum + scatter:
//   out[w,i] = sum_{j,l} tt[w,j]*xq[w,l]*W_bil[i,j,l] -> atomicAdd agg[idx_ji[w],i]
// Wbf: fragment-ordered bf16 (512 KB). Per 32-K chunk its 4 KB tile is
// staged to LDS with global_load_async_to_lds_b128 (ASYNCcnt path).
// Each lane register-caches the 32 xq values it ever needs.
// =====================================================================
__global__ __launch_bounds__(256) void bilinear_scatter(
    const float* __restrict__ TT,  // [T,64]
    const float* __restrict__ XQ,  // [T,64]
    const __bf16* __restrict__ Wbf,
    const int*   __restrict__ idx_ji,
    float* __restrict__ agg)       // [E,64]
{
    __shared__ __bf16 Ws[2048];    // one 32x64 fragment-ordered tile (4 KB)

    const int tid  = threadIdx.x;
    const int wave = tid >> 5;
    const int lane = tid & 31;
    const size_t r0   = (size_t)blockIdx.x * 128;
    const size_t rowl = r0 + (wave << 4) + (lane & 15);
    const int half = (lane >> 4) << 3;

    // register-cache xq runs: run g covers cols g*16+half .. +8
    float xq[4][8];
    {
        const float* xr = XQ + rowl * 64;
#pragma unroll
        for (int g = 0; g < 4; ++g) {
            float4 q0 = *(const float4*)(xr + g * 16 + half);
            float4 q1 = *(const float4*)(xr + g * 16 + half + 4);
            xq[g][0] = q0.x; xq[g][1] = q0.y; xq[g][2] = q0.z; xq[g][3] = q0.w;
            xq[g][4] = q1.x; xq[g][5] = q1.y; xq[g][6] = q1.z; xq[g][7] = q1.w;
        }
    }
    const float* tr = TT + rowl * 64;

    v8f acc[4] = {};

    for (int kc = 0; kc < 128; ++kc) {
        __syncthreads();
        {
            const uint4* gsrc = (const uint4*)Wbf + (size_t)kc * 256 + tid;
#if defined(__gfx1250__)
            unsigned int ldsoff = (unsigned int)(size_t)(&Ws[tid * 8]);
            asm volatile("global_load_async_to_lds_b128 %0, %1, off"
                         :: "v"(ldsoff), "v"((unsigned long long)(size_t)gsrc)
                         : "memory");
            asm volatile("s_wait_asynccnt 0x0" ::: "memory");
#else
            *(uint4*)(&Ws[tid * 8]) = *gsrc;
#endif
        }
        __syncthreads();

        const int j   = kc >> 1;
        const int sel = (kc & 1) << 1;        // lb=0 -> runs 0,1 ; lb=32 -> runs 2,3
        const float ts = tr[j];
        v16bf a;
#pragma unroll
        for (int q = 0; q < 8; ++q) a[q]     = (__bf16)(ts * xq[sel][q]);
#pragma unroll
        for (int q = 0; q < 8; ++q) a[8 + q] = (__bf16)(ts * xq[sel + 1][q]);

#pragma unroll
        for (int ct = 0; ct < 4; ++ct) {
            v16bf b = *(const v16bf*)(&Ws[(ct << 9) + (lane << 4)]);
            acc[ct] = __builtin_amdgcn_wmma_f32_16x16x32_bf16(
                false, a, false, b, (short)0, acc[ct], false, false);
        }
    }

    const int hi = (lane >> 4) << 3;
#pragma unroll
    for (int r = 0; r < 8; ++r) {
        const size_t w = r0 + (wave << 4) + hi + r;
        const int dst = idx_ji[w];
#pragma unroll
        for (int ct = 0; ct < 4; ++ct) {
            const int col = (ct << 4) + (lane & 15);
            atomAddF(&agg[(size_t)dst * 64 + col], acc[ct][r]);
        }
    }
}

// ---------------- small VALU kernels ----------------

__global__ void rbf_proj(const float* __restrict__ rbf, const float* __restrict__ W1,
                         const float* __restrict__ W2, const float* __restrict__ mul,
                         float* __restrict__ out)
{
    const size_t gid = (size_t)blockIdx.x * 256 + threadIdx.x;
    const size_t e = gid >> 7;
    const int h = (int)(gid & 127);
    float r[6];
#pragma unroll
    for (int i = 0; i < 6; ++i) r[i] = rbf[e * 6 + i];
    float acc = 0.0f;
#pragma unroll
    for (int b = 0; b < 8; ++b) {
        float t = 0.0f;
#pragma unroll
        for (int i = 0; i < 6; ++i) t += r[i] * W1[i * 8 + b];
        acc += t * W2[b * 128 + h];
    }
    if (mul) acc *= mul[gid];
    out[gid] = acc;
}

__global__ void rbf_out_mul(const float* __restrict__ rbf, const float* __restrict__ Wr,
                            const float* __restrict__ e1, float* __restrict__ e2)
{
    const size_t gid = (size_t)blockIdx.x * 256 + threadIdx.x;
    const size_t e = gid >> 7;
    const int h = (int)(gid & 127);
    float acc = 0.0f;
#pragma unroll
    for (int i = 0; i < 6; ++i) acc += rbf[e * 6 + i] * Wr[i * 128 + h];
    e2[gid] = acc * e1[gid];
}

__global__ void sbf_stage1(const float* __restrict__ sbf, const float* __restrict__ W1,
                           float* __restrict__ U)
{
    const size_t gid = (size_t)blockIdx.x * 256 + threadIdx.x;
    const size_t w = gid >> 3;
    const int a = (int)(gid & 7);
    float acc = 0.0f;
#pragma unroll
    for (int s = 0; s < 42; ++s) acc += sbf[w * 42 + s] * W1[s * 8 + a];
    U[gid] = acc;
}

__global__ void xqt_build(const float* __restrict__ U, const float* __restrict__ W2,
                          const float* __restrict__ xqd, const int* __restrict__ idx_kj,
                          float* __restrict__ XQT)
{
    const size_t gid = (size_t)blockIdx.x * 256 + threadIdx.x;
    const size_t w = gid >> 6;
    const int l = (int)(gid & 63);
    float sb = 0.0f;
#pragma unroll
    for (int a = 0; a < 8; ++a) sb += U[w * 8 + a] * W2[a * 64 + l];
    XQT[gid] = xqd[(size_t)idx_kj[w] * 64 + l] * sb;
}

__global__ void gather_scatter64(const float* __restrict__ src, const int* __restrict__ idx_kj,
                                 const int* __restrict__ idx_ji, float* __restrict__ dst)
{
    const size_t gid = (size_t)blockIdx.x * 256 + threadIdx.x;
    const size_t w = gid >> 6;
    const int c = (int)(gid & 63);
    const float v = src[(size_t)idx_kj[w] * 64 + c];
    atomAddF(&dst[(size_t)idx_ji[w] * 64 + c], v);
}

__global__ void add_ew(const float* __restrict__ a, const float* __restrict__ b,
                       float* __restrict__ c)
{
    const size_t gid = (size_t)blockIdx.x * 256 + threadIdx.x;
    c[gid] = a[gid] + b[gid];
}

__global__ void zero_f32(float* __restrict__ p)
{
    p[(size_t)blockIdx.x * 256 + threadIdx.x] = 0.0f;
}

// =====================================================================
extern "C" void kernel_launch(void* const* d_in, const int* in_sizes, int n_in,
                              void* d_out, int out_size, void* d_ws, size_t ws_size,
                              hipStream_t stream)
{
    const float* x1      = (const float*)d_in[0];
    const float* rbf0    = (const float*)d_in[1];
    const float* sbf     = (const float*)d_in[2];
    const float* t_in    = (const float*)d_in[3];
    const float* rbf0_g  = (const float*)d_in[4];
    const int*   idx_kj  = (const int*)d_in[5];
    const int*   idx_ji  = (const int*)d_in[6];
    const float* W_ji    = (const float*)d_in[7];
    const float* b_ji    = (const float*)d_in[8];
    const float* W_kj    = (const float*)d_in[9];
    const float* b_kj    = (const float*)d_in[10];
    const float* W_rbf1g = (const float*)d_in[11];
    const float* W_rbf2g = (const float*)d_in[12];
    const float* W_downg = (const float*)d_in[13];
    const float* W_upg   = (const float*)d_in[14];
    const float* W_rbf1  = (const float*)d_in[15];
    const float* W_rbf2  = (const float*)d_in[16];
    const float* W_sbf1  = (const float*)d_in[17];
    const float* W_sbf2  = (const float*)d_in[18];
    const float* W_t1    = (const float*)d_in[19];
    const float* W_downq = (const float*)d_in[20];
    const float* W_upq   = (const float*)d_in[21];
    const float* W_bil   = (const float*)d_in[22];
    const float* W_skip  = (const float*)d_in[23];
    const float* b_skip  = (const float*)d_in[24];
    const float* W_rbfo  = (const float*)d_in[25];
    const float* rb_w1   = (const float*)d_in[26];
    const float* rb_b1   = (const float*)d_in[27];
    const float* rb_w2   = (const float*)d_in[28];
    const float* rb_b2   = (const float*)d_in[29];
    const float* ra1_w1  = (const float*)d_in[30];
    const float* ra1_b1  = (const float*)d_in[31];
    const float* ra1_w2  = (const float*)d_in[32];
    const float* ra1_b2  = (const float*)d_in[33];
    const float* ra2_w1  = (const float*)d_in[34];
    const float* ra2_b1  = (const float*)d_in[35];
    const float* ra2_w2  = (const float*)d_in[36];
    const float* ra2_b2  = (const float*)d_in[37];

    const size_t EH  = (size_t)NEDGE * 128;
    const size_t E64 = (size_t)NEDGE * 64;
    const size_t T64 = (size_t)NTRIP * 64;

    // ---- f32 workspace ----
    float* ws  = (float*)d_ws;
    float* B1  = ws;             // g buffers / scratch
    float* B2  = ws + 1 * EH;    // x_ji_g
    float* B3  = ws + 2 * EH;    // x_kj_g
    float* B4  = ws + 3 * EH;    // scratch
    float* B5  = ws + 4 * EH;    // qmpg
    float* D1  = ws + 5 * EH;    // E x 64
    float* D2  = D1 + E64;       // E x 64
    float* TT  = D2 + E64;       // T x 64
    float* XQT = TT + T64;       // T x 64
    float* U   = XQT + T64;      // T x 8
    const size_t F32TOTAL = 5 * EH + 2 * E64 + 2 * T64 + (size_t)NTRIP * 8;

    // ---- bf16 workspace (fragment-ordered weights + padded t) ----
    __bf16* bf = (__bf16*)(ws + F32TOTAL);
    const size_t SW_HH = 16384;  // [128,128]
    __bf16* SWji   = bf + 0 * SW_HH;
    __bf16* SWkj   = bf + 1 * SW_HH;
    __bf16* SWskip = bf + 2 * SW_HH;
    __bf16* SWrb1  = bf + 3 * SW_HH;
    __bf16* SWrb2  = bf + 4 * SW_HH;
    __bf16* SWa11  = bf + 5 * SW_HH;
    __bf16* SWa12  = bf + 6 * SW_HH;
    __bf16* SWa21  = bf + 7 * SW_HH;
    __bf16* SWa22  = bf + 8 * SW_HH;
    __bf16* SWdng  = bf + 9 * SW_HH;              // [128,64] = 8192
    __bf16* SWdnq  = SWdng + 8192;
    __bf16* SWupg  = SWdnq + 8192;                // [64,128] = 8192
    __bf16* SWupq  = SWupg + 8192;
    __bf16* SWt1   = SWupq + 8192;                // [294->320,64] = 20480
    __bf16* SWbil  = SWt1 + 20480;                // 262144
    __bf16* TB     = SWbil + 262144;              // t padded bf16 [T,320]

    const int GEB  = NEDGE / 128;
    const int GTB  = NTRIP / 128;
    const int GEH  = (int)(EH / 256);
    const int GE64 = (int)(E64 / 256);
    const int GT64 = (int)(T64 / 256);
    const int GT8  = (int)((size_t)NTRIP * 8 / 256);

    const int BS  = MODE_BIAS | MODE_SWISH;
    const int BSM = BS | MODE_MUL;
    const int BSA = BS | MODE_ADD;

    // ---------- pre-swizzle weights ----------
    swizzle_w<<<4, 256, 0, stream>>>(W_ji,   SWji,   128, 128, 4, 8);
    swizzle_w<<<4, 256, 0, stream>>>(W_kj,   SWkj,   128, 128, 4, 8);
    swizzle_w<<<4, 256, 0, stream>>>(W_skip, SWskip, 128, 128, 4, 8);
    swizzle_w<<<4, 256, 0, stream>>>(rb_w1,  SWrb1,  128, 128, 4, 8);
    swizzle_w<<<4, 256, 0, stream>>>(rb_w2,  SWrb2,  128, 128, 4, 8);
    swizzle_w<<<4, 256, 0, stream>>>(ra1_w1, SWa11,  128, 128, 4, 8);
    swizzle_w<<<4, 256, 0, stream>>>(ra1_w2, SWa12,  128, 128, 4, 8);
    swizzle_w<<<4, 256, 0, stream>>>(ra2_w1, SWa21,  128, 128, 4, 8);
    swizzle_w<<<4, 256, 0, stream>>>(ra2_w2, SWa22,  128, 128, 4, 8);
    swizzle_w<<<2, 256, 0, stream>>>(W_downg, SWdng, 128, 64, 4, 4);
    swizzle_w<<<2, 256, 0, stream>>>(W_downq, SWdnq, 128, 64, 4, 4);
    swizzle_w<<<2, 256, 0, stream>>>(W_upg,   SWupg, 64, 128, 2, 8);
    swizzle_w<<<2, 256, 0, stream>>>(W_upq,   SWupq, 64, 128, 2, 8);
    swizzle_w<<<5, 256, 0, stream>>>(W_t1,    SWt1,  294, 64, 10, 4);
    swizzle_bil<<<64, 256, 0, stream>>>(W_bil, SWbil);
    cvt_pad_t<<<NTRIP, 256, 0, stream>>>(t_in, TB);

    // ---------- update_eG ----------
    rbf_proj<<<GEH, 256, 0, stream>>>(rbf0_g, W_rbf1g, W_rbf2g, nullptr, B1);
    gemm2<128, false><<<GEB, 256, 0, stream>>>(x1, SWji, b_ji, nullptr, nullptr, B2, 128, BS);
    gemm2<128, false><<<GEB, 256, 0, stream>>>(x1, SWkj, b_kj, B1, nullptr, B4, 128, BSM);
    gemm2<64, false><<<GEB, 256, 0, stream>>>(B4, SWdng, nullptr, nullptr, nullptr, D1, 128, MODE_SWISH);
    zero_f32<<<GE64, 256, 0, stream>>>(D2);
    gather_scatter64<<<GT64, 256, 0, stream>>>(D1, idx_kj, idx_ji, D2);
    gemm2<128, false><<<GEB, 256, 0, stream>>>(D2, SWupg, nullptr, nullptr, nullptr, B3, 64, MODE_SWISH);
    add_ew<<<GEH, 256, 0, stream>>>(B2, B3, B5);
    gemm2<128, false><<<GEB, 256, 0, stream>>>(B5, SWrb1, rb_b1, nullptr, nullptr, B4, 128, BS);
    gemm2<128, false><<<GEB, 256, 0, stream>>>(B4, SWrb2, rb_b2, nullptr, B5, B5, 128, BSA);
    gemm2<128, false><<<GEB, 256, 0, stream>>>(B5, SWskip, b_skip, nullptr, x1, B5, 128, BSA);
    gemm2<128, false><<<GEB, 256, 0, stream>>>(B5, SWa11, ra1_b1, nullptr, nullptr, B4, 128, BS);
    gemm2<128, false><<<GEB, 256, 0, stream>>>(B4, SWa12, ra1_b2, nullptr, B5, B5, 128, BSA);
    gemm2<128, false><<<GEB, 256, 0, stream>>>(B5, SWa21, ra2_b1, nullptr, nullptr, B4, 128, BS);
    gemm2<128, false><<<GEB, 256, 0, stream>>>(B4, SWa22, ra2_b2, nullptr, B5, B5, 128, BSA);

    // ---------- update_eQ ----------
    rbf_proj<<<GEH, 256, 0, stream>>>(rbf0, W_rbf1, W_rbf2, B3, B4);   // x_kj_g * g_q
    gemm2<64, false><<<GEB, 256, 0, stream>>>(B4, SWdnq, nullptr, nullptr, nullptr, D1, 128, MODE_SWISH);
    sbf_stage1<<<GT8, 256, 0, stream>>>(sbf, W_sbf1, U);
    xqt_build<<<GT64, 256, 0, stream>>>(U, W_sbf2, D1, idx_kj, XQT);
    gemm2<64, true><<<GTB, 256, 0, stream>>>(TB, SWt1, nullptr, nullptr, nullptr, TT, 320, 0);
    zero_f32<<<GE64, 256, 0, stream>>>(D2);
    bilinear_scatter<<<GTB, 256, 0, stream>>>(TT, XQT, SWbil, idx_ji, D2);

    float* OUT1 = (float*)d_out;
    float* OUT2 = OUT1 + EH;
    gemm2<128, false><<<GEB, 256, 0, stream>>>(D2, SWupq, nullptr, nullptr, B2, OUT1, 64, MODE_SWISH | MODE_ADD);
    gemm2<128, false><<<GEB, 256, 0, stream>>>(OUT1, SWrb1, rb_b1, nullptr, nullptr, B4, 128, BS);
    gemm2<128, false><<<GEB, 256, 0, stream>>>(B4, SWrb2, rb_b2, nullptr, OUT1, OUT1, 128, BSA);
    gemm2<128, false><<<GEB, 256, 0, stream>>>(OUT1, SWskip, b_skip, nullptr, B5, OUT1, 128, BSA);
    gemm2<128, false><<<GEB, 256, 0, stream>>>(OUT1, SWa11, ra1_b1, nullptr, nullptr, B4, 128, BS);
    gemm2<128, false><<<GEB, 256, 0, stream>>>(B4, SWa12, ra1_b2, nullptr, OUT1, OUT1, 128, BSA);
    gemm2<128, false><<<GEB, 256, 0, stream>>>(OUT1, SWa21, ra2_b1, nullptr, nullptr, B4, 128, BS);
    gemm2<128, false><<<GEB, 256, 0, stream>>>(B4, SWa22, ra2_b2, nullptr, OUT1, OUT1, 128, BSA);

    // ---------- e2 ----------
    rbf_out_mul<<<GEH, 256, 0, stream>>>(rbf0, W_rbfo, OUT1, OUT2);
}
